// SpatialTransformer_13967233647516
// MI455X (gfx1250) — compile-verified, matrix operand order checked by reading, not run on hardware
//
#include <hip/hip_runtime.h>
#include <hip/hip_bf16.h>
#include <math.h>

// ---------------- problem constants ----------------
#define Bsz     8
#define Cdim    320
#define HWdim   1024
#define HEADSn  8
#define DHEADn  40
#define CTXLEN  77
#define CTXDIM  768
#define FFIN    1280      // FF_INNER
#define GROUPSn 32
#define MTOK    (Bsz * HWdim)    // 8192
#define MCTX    (Bsz * CTXLEN)   // 616

// ---------------- WMMA plumbing (gfx1250 wave32) ----------------
typedef __attribute__((ext_vector_type(16))) __bf16 v16bf;
typedef __attribute__((ext_vector_type(8)))  float  v8f;

union Frag16 { v16bf v; uint4 q[2]; unsigned short u[16]; };

__device__ __forceinline__ unsigned short f2bf(float f) {
  union { float f; unsigned u; } a; a.f = f;
  unsigned r = a.u + 0x7FFFu + ((a.u >> 16) & 1u);   // RNE
  return (unsigned short)(r >> 16);
}

__device__ __forceinline__ v8f wmma_bf16(const Frag16& a, const Frag16& b, v8f c) {
  // D = A(16x32 bf16) * B(32x16 bf16) + C(16x16 f32)
  return __builtin_amdgcn_wmma_f32_16x16x32_bf16(false, a.v, false, b.v,
                                                 (short)0, c, false, false);
}

// ================= one-time converters =================
// W[K,N] f32  ->  Wt[N,K] bf16 (transposed for LDS-friendly B tiles)
__global__ __launch_bounds__(256)
void wt_cvt_kernel(const float* __restrict__ W, unsigned short* __restrict__ Wt,
                   int K, int N) {
  int idx = blockIdx.x * 256 + threadIdx.x;
  if (idx >= K * N) return;
  int k = idx / N, n = idx % N;
  Wt[(size_t)n * K + k] = f2bf(W[idx]);
}

__global__ __launch_bounds__(256)
void act_cvt_kernel(const float* __restrict__ X, unsigned short* __restrict__ Y, int n) {
  int idx = blockIdx.x * 256 + threadIdx.x;
  if (idx < n) Y[idx] = f2bf(X[idx]);
}

// ================= GroupNorm =================
__global__ __launch_bounds__(256)
void gn_stats_kernel(const float* __restrict__ x, float* __restrict__ stats) {
  const int bg = blockIdx.x;                       // b*GROUPS + g
  const int n  = (Cdim / GROUPSn) * HWdim;         // 10240 contiguous floats
  const float* base = x + (size_t)bg * n;
  float s1 = 0.f, s2 = 0.f;
  for (int i = threadIdx.x; i < n; i += 256) { float v = base[i]; s1 += v; s2 += v * v; }
  const int lane = threadIdx.x & 31, wid = threadIdx.x >> 5;
#pragma unroll
  for (int off = 16; off >= 1; off >>= 1) {
    s1 += __shfl_xor(s1, off, 32);
    s2 += __shfl_xor(s2, off, 32);
  }
  __shared__ float r1[8], r2[8];
  if (lane == 0) { r1[wid] = s1; r2[wid] = s2; }
  __syncthreads();
  if (threadIdx.x == 0) {
    float S1 = 0.f, S2 = 0.f;
#pragma unroll
    for (int i = 0; i < 8; ++i) { S1 += r1[i]; S2 += r2[i]; }
    float mu  = S1 / (float)n;
    float var = S2 / (float)n - mu * mu;
    stats[2 * bg]     = mu;
    stats[2 * bg + 1] = rsqrtf(var + 1e-6f);
  }
}

__global__ __launch_bounds__(256)
void gn_apply_kernel(const float* __restrict__ x, const float* __restrict__ stats,
                     const float* __restrict__ sc, const float* __restrict__ bi,
                     unsigned short* __restrict__ out) {  // out: [B, HW, C] bf16
  int idx = blockIdx.x * 256 + threadIdx.x;
  if (idx >= Bsz * Cdim * HWdim) return;
  int p = idx & (HWdim - 1);
  int c = (idx >> 10) % Cdim;
  int b = idx / (Cdim * HWdim);
  int gidx = b * GROUPSn + c / (Cdim / GROUPSn);
  float mu = stats[2 * gidx], rs = stats[2 * gidx + 1];
  out[((size_t)b * HWdim + p) * Cdim + c] = f2bf((x[idx] - mu) * rs * sc[c] + bi[c]);
}

// ================= LayerNorm (one wave per 320-wide row), bf16 out =================
__global__ __launch_bounds__(256)
void layernorm_kernel(const float* __restrict__ x, const float* __restrict__ s,
                      const float* __restrict__ bb, unsigned short* __restrict__ y, int M) {
  const int wid = threadIdx.x >> 5, lane = threadIdx.x & 31;
  const int row = blockIdx.x * 8 + wid;
  if (row >= M) return;
  const float* xr = x + (size_t)row * Cdim;
  float v[10], s1 = 0.f, s2 = 0.f;
#pragma unroll
  for (int i = 0; i < 10; ++i) { v[i] = xr[lane + i * 32]; s1 += v[i]; s2 += v[i] * v[i]; }
#pragma unroll
  for (int off = 16; off >= 1; off >>= 1) {
    s1 += __shfl_xor(s1, off, 32);
    s2 += __shfl_xor(s2, off, 32);
  }
  float mu = s1 * (1.f / Cdim);
  float rs = rsqrtf(s2 * (1.f / Cdim) - mu * mu + 1e-5f);
  unsigned short* yr = y + (size_t)row * Cdim;
#pragma unroll
  for (int i = 0; i < 10; ++i) {
    int c = lane + i * 32;
    yr[c] = f2bf((v[i] - mu) * rs * s[c] + bb[c]);
  }
}

// ========== bf16-WMMA GEMM: C = A[M,K]bf16 @ Bt[N,K]bf16^T (+bias)(+res) ==========
// block: 256 threads = 8 waves; tile 128x64, k-step 32; wave -> 32x32 (2x2 wmma tiles)
// Pure b128 staging: no conversions in the hot loop.
template <int EPI, int OBF>   // EPI 0:none 1:+bias 2:+bias+res ; OBF 0:f32 out 1:bf16 out
__global__ __launch_bounds__(256)
void gemm_bf16_kernel(const unsigned short* __restrict__ A,
                      const unsigned short* __restrict__ Bt,
                      const float* __restrict__ bias, const float* __restrict__ res,
                      void* __restrict__ Cout, int M, int N, int K) {
  __shared__ unsigned short sA[128][40];   // [m][k], pad 32->40
  __shared__ unsigned short sBT[64][40];   // [n][k]
  const int tid = threadIdx.x;
  const int lane = tid & 31, wid = tid >> 5;
  const int wr = wid >> 1, wc = wid & 1;
  const int halfq = lane >> 4, l15 = lane & 15;
  const int m0 = blockIdx.y * 128, n0 = blockIdx.x * 64;

  const v8f zero8 = {0.f, 0.f, 0.f, 0.f, 0.f, 0.f, 0.f, 0.f};
  v8f acc[2][2] = {{zero8, zero8}, {zero8, zero8}};

  const int arow = tid >> 1, acb = (tid & 1) * 16;   // 2 thr/row x 16 bf16
  const int bn = tid >> 2, bqo = (tid & 3) * 8;      // 4 thr/row x 8 bf16
  const int nk = K >> 5;

  for (int kt = 0; kt < nk; ++kt) {
    { // stage A tile (raw bf16 b128 copies)
      const int gm = m0 + arow;
      const unsigned short* ap = A + (size_t)gm * K + (kt << 5) + acb;
      uint4 v0, v1;
      if (gm < M) { v0 = *(const uint4*)ap; v1 = *(const uint4*)(ap + 8); }
      else        { v0 = make_uint4(0u, 0u, 0u, 0u); v1 = v0; }
      *(uint4*)&sA[arow][acb]     = v0;
      *(uint4*)&sA[arow][acb + 8] = v1;
      if (gm < M && kt + 1 < nk) __builtin_prefetch(ap + 32, 0, 0);
    }
    { // stage B tile (already [n][k] in global: straight b128 copy)
      const unsigned short* bp = Bt + (size_t)(n0 + bn) * K + (kt << 5) + bqo;
      *(uint4*)&sBT[bn][bqo] = *(const uint4*)bp;
      if (kt + 1 < nk) __builtin_prefetch(bp + 32, 0, 0);
    }
    __syncthreads();

    Frag16 fa[2], fb[2];
#pragma unroll
    for (int mt = 0; mt < 2; ++mt) {
      const unsigned short* p = &sA[wr * 32 + mt * 16 + l15][halfq * 8];
      fa[mt].q[0] = *(const uint4*)p;          // k = koff .. koff+7
      fa[mt].q[1] = *(const uint4*)(p + 16);   // k = koff+16 .. koff+23
    }
#pragma unroll
    for (int nt = 0; nt < 2; ++nt) {
      const unsigned short* p = &sBT[wc * 32 + nt * 16 + l15][halfq * 8];
      fb[nt].q[0] = *(const uint4*)p;
      fb[nt].q[1] = *(const uint4*)(p + 16);
    }
#pragma unroll
    for (int mt = 0; mt < 2; ++mt)
#pragma unroll
      for (int nt = 0; nt < 2; ++nt)
        acc[mt][nt] = wmma_bf16(fa[mt], fb[nt], acc[mt][nt]);
    __syncthreads();
  }

#pragma unroll
  for (int mt = 0; mt < 2; ++mt)
#pragma unroll
    for (int nt = 0; nt < 2; ++nt)
#pragma unroll
      for (int r = 0; r < 8; ++r) {
        int m = m0 + wr * 32 + mt * 16 + r + halfq * 8;
        int n = n0 + wc * 32 + nt * 16 + l15;
        if (m < M) {
          float v = acc[mt][nt][r];
          if (EPI >= 1) v += bias[n];
          if (EPI == 2) v += res[(size_t)m * N + n];
          if (OBF) ((unsigned short*)Cout)[(size_t)m * N + n] = f2bf(v);
          else     ((float*)Cout)[(size_t)m * N + n] = v;
        }
      }
}

// ====== fused GEGLU GEMM: act = (x@W1t_a + b1a) * gelu(x@W1t_g + b1g), bf16 out ======
// W1t: [2560, 320] bf16 (pre-transposed); rows [0,1280) = a-half, [1280,2560) = g-half
__global__ __launch_bounds__(256)
void geglu_kernel(const unsigned short* __restrict__ A,
                  const unsigned short* __restrict__ W1t,
                  const float* __restrict__ b1, unsigned short* __restrict__ act, int M) {
  __shared__ unsigned short sA[128][40];
  __shared__ unsigned short sBa[64][40];
  __shared__ unsigned short sBg[64][40];
  const int tid = threadIdx.x;
  const int lane = tid & 31, wid = tid >> 5;
  const int wr = wid >> 1, wc = wid & 1;
  const int halfq = lane >> 4, l15 = lane & 15;
  const int m0 = blockIdx.y * 128, n0 = blockIdx.x * 64;

  const v8f zero8 = {0.f, 0.f, 0.f, 0.f, 0.f, 0.f, 0.f, 0.f};
  v8f aa[2][2] = {{zero8, zero8}, {zero8, zero8}};
  v8f ag[2][2] = {{zero8, zero8}, {zero8, zero8}};

  const int arow = tid >> 1, acb = (tid & 1) * 16;
  const int bn = tid >> 2, bqo = (tid & 3) * 8;
  const int nk = Cdim >> 5;  // 10

  for (int kt = 0; kt < nk; ++kt) {
    {
      const int gm = m0 + arow;
      const unsigned short* ap = A + (size_t)gm * Cdim + (kt << 5) + acb;
      uint4 v0, v1;
      if (gm < M) { v0 = *(const uint4*)ap; v1 = *(const uint4*)(ap + 8); }
      else        { v0 = make_uint4(0u, 0u, 0u, 0u); v1 = v0; }
      *(uint4*)&sA[arow][acb]     = v0;
      *(uint4*)&sA[arow][acb + 8] = v1;
    }
    {
      const unsigned short* pa = W1t + (size_t)(n0 + bn) * Cdim + (kt << 5) + bqo;
      const unsigned short* pg = pa + (size_t)FFIN * Cdim;
      *(uint4*)&sBa[bn][bqo] = *(const uint4*)pa;
      *(uint4*)&sBg[bn][bqo] = *(const uint4*)pg;
    }
    __syncthreads();

    Frag16 fa[2], fba[2], fbg[2];
#pragma unroll
    for (int mt = 0; mt < 2; ++mt) {
      const unsigned short* p = &sA[wr * 32 + mt * 16 + l15][halfq * 8];
      fa[mt].q[0] = *(const uint4*)p; fa[mt].q[1] = *(const uint4*)(p + 16);
    }
#pragma unroll
    for (int nt = 0; nt < 2; ++nt) {
      const unsigned short* pa = &sBa[wc * 32 + nt * 16 + l15][halfq * 8];
      const unsigned short* pg = &sBg[wc * 32 + nt * 16 + l15][halfq * 8];
      fba[nt].q[0] = *(const uint4*)pa; fba[nt].q[1] = *(const uint4*)(pa + 16);
      fbg[nt].q[0] = *(const uint4*)pg; fbg[nt].q[1] = *(const uint4*)(pg + 16);
    }
#pragma unroll
    for (int mt = 0; mt < 2; ++mt)
#pragma unroll
      for (int nt = 0; nt < 2; ++nt) {
        aa[mt][nt] = wmma_bf16(fa[mt], fba[nt], aa[mt][nt]);
        ag[mt][nt] = wmma_bf16(fa[mt], fbg[nt], ag[mt][nt]);
      }
    __syncthreads();
  }

#pragma unroll
  for (int mt = 0; mt < 2; ++mt)
#pragma unroll
    for (int nt = 0; nt < 2; ++nt)
#pragma unroll
      for (int r = 0; r < 8; ++r) {
        int m = m0 + wr * 32 + mt * 16 + r + halfq * 8;
        int n = n0 + wc * 32 + nt * 16 + l15;
        if (m < M) {
          float a = aa[mt][nt][r] + b1[n];
          float g = ag[mt][nt][r] + b1[FFIN + n];
          float gl = 0.5f * g * (1.f + erff(g * 0.70710678118654752f));  // exact gelu
          act[(size_t)m * FFIN + n] = f2bf(a * gl);
        }
      }
}

// ========== flash attention: 4 waves/block, one (b, head, 16-query tile) per wave ==========
// Q,K,V,O bf16: [rows, 320] (head h in cols h*40..h*40+39); mask: [B, HW, Lk] bytes.
// Key tile = 32, d padded 40->64 (A side) / 48 (V columns).
__global__ __launch_bounds__(128)
void attn_kernel(const unsigned short* __restrict__ Q, const unsigned short* __restrict__ Kg,
                 const unsigned short* __restrict__ Vg, const unsigned char* __restrict__ mask,
                 unsigned short* __restrict__ O, int Lk, float scale) {
  const int tid = threadIdx.x;
  const int wv = tid >> 5, lane = tid & 31;
  const int qt = blockIdx.x * 4 + wv;
  const int bh = blockIdx.y;
  const int b = bh >> 3, h = bh & 7;
  const int halfq = lane >> 4, l15 = lane & 15;
  const int q0 = qt << 4;
  const size_t qrow = (size_t)b * HWdim + q0;
  const int dc = h * DHEADn;

  __shared__ unsigned short sK[4][32][72];    // [wave][key][d 0..63 pad]
  __shared__ unsigned short sVT[4][48][40];   // [wave][d][key 0..31 pad]
  __shared__ unsigned short sP[4][16][40];    // [wave][qrow][key 0..31 pad]

  // Q fragments for padded d=64 (2 k-steps) -- raw bf16, no conversion
  Frag16 aq[2];
#pragma unroll
  for (int kt = 0; kt < 2; ++kt)
#pragma unroll
    for (int j = 0; j < 16; ++j) {
      int d = kt * 32 + (j & 7) + ((j >> 3) << 4) + (halfq << 3);
      aq[kt].u[j] = (d < DHEADn) ? Q[(qrow + l15) * Cdim + dc + d] : (unsigned short)0;
    }

  const v8f zero8 = {0.f, 0.f, 0.f, 0.f, 0.f, 0.f, 0.f, 0.f};
  v8f o[3] = {zero8, zero8, zero8};
  float m_r[8], l_r[8];
#pragma unroll
  for (int r = 0; r < 8; ++r) { m_r[r] = -1e30f; l_r[r] = 0.f; }

  const int nkt = (Lk + 31) >> 5;
  for (int jt = 0; jt < nkt; ++jt) {
    const int kb = jt << 5;
    const int key = kb + lane;
    const bool kvalid = key < Lk;
    const unsigned short* kp = Kg + ((size_t)b * Lk + key) * Cdim + dc;
    const unsigned short* vp = Vg + ((size_t)b * Lk + key) * Cdim + dc;
#pragma unroll
    for (int d = 0; d < 64; d += 2)   // pairwise u32 copies (dc, d both even)
      *(unsigned*)&sK[wv][lane][d] =
          (kvalid && d < DHEADn) ? *(const unsigned*)(kp + d) : 0u;
#pragma unroll
    for (int d = 0; d < 48; ++d)
      sVT[wv][d][lane] = (kvalid && d < DHEADn) ? vp[d] : (unsigned short)0;
    __syncthreads();

    // S[16, 32] = Q @ K^T (two key-column tiles, two k-steps each)
    Frag16 bk[2][2];
#pragma unroll
    for (int ct = 0; ct < 2; ++ct)
#pragma unroll
      for (int kt = 0; kt < 2; ++kt) {
        const unsigned short* p = &sK[wv][ct * 16 + l15][kt * 32 + halfq * 8];
        bk[kt][ct].q[0] = *(const uint4*)p;
        bk[kt][ct].q[1] = *(const uint4*)(p + 16);
      }
    v8f s0 = zero8, s1 = zero8;
    s0 = wmma_bf16(aq[0], bk[0][0], s0);
    s0 = wmma_bf16(aq[1], bk[1][0], s0);
    s1 = wmma_bf16(aq[0], bk[0][1], s1);
    s1 = wmma_bf16(aq[1], bk[1][1], s1);

    // scale + mask + online softmax (C-frag row M = r + 8*halfq, col N = l15)
    float pv0[8], pv1[8], tmax[8];
#pragma unroll
    for (int r = 0; r < 8; ++r) {
      int qr = q0 + r + halfq * 8;
      const unsigned char* mrow = mask + ((size_t)b * HWdim + qr) * Lk;
      int k0c = kb + l15, k1c = kb + 16 + l15;
      float v0 = s0[r] * scale;
      float v1 = s1[r] * scale;
      v0 = (k0c < Lk && mrow[k0c]) ? v0 : -1e30f;
      v1 = (k1c < Lk && mrow[k1c]) ? v1 : -1e30f;
      pv0[r] = v0; pv1[r] = v1;
      tmax[r] = fmaxf(v0, v1);
    }
#pragma unroll
    for (int off = 1; off < 16; off <<= 1)
#pragma unroll
      for (int r = 0; r < 8; ++r)
        tmax[r] = fmaxf(tmax[r], __shfl_xor(tmax[r], off, 32));

    float alpha[8], rsum[8];
#pragma unroll
    for (int r = 0; r < 8; ++r) {
      float mn = fmaxf(m_r[r], tmax[r]);
      alpha[r] = __expf(m_r[r] - mn);
      m_r[r] = mn;
      pv0[r] = __expf(pv0[r] - mn);
      pv1[r] = __expf(pv1[r] - mn);
      rsum[r] = pv0[r] + pv1[r];
    }
#pragma unroll
    for (int off = 1; off < 16; off <<= 1)
#pragma unroll
      for (int r = 0; r < 8; ++r)
        rsum[r] += __shfl_xor(rsum[r], off, 32);
#pragma unroll
    for (int r = 0; r < 8; ++r) l_r[r] = l_r[r] * alpha[r] + rsum[r];
#pragma unroll
    for (int nt = 0; nt < 3; ++nt)
#pragma unroll
      for (int r = 0; r < 8; ++r) o[nt][r] *= alpha[r];

    // write P (C-frag layout) into LDS, reread as A fragment
#pragma unroll
    for (int r = 0; r < 8; ++r) {
      sP[wv][r + halfq * 8][l15]      = f2bf(pv0[r]);
      sP[wv][r + halfq * 8][16 + l15] = f2bf(pv1[r]);
    }
    __syncthreads();

    Frag16 fp;
    {
      const unsigned short* p = &sP[wv][l15][halfq * 8];
      fp.q[0] = *(const uint4*)p;
      fp.q[1] = *(const uint4*)(p + 16);
    }
#pragma unroll
    for (int nt = 0; nt < 3; ++nt) {
      Frag16 bv;
      const unsigned short* p = &sVT[wv][nt * 16 + l15][halfq * 8];
      bv.q[0] = *(const uint4*)p;
      bv.q[1] = *(const uint4*)(p + 16);
      o[nt] = wmma_bf16(fp, bv, o[nt]);   // O += P[16,32] @ V[32,16]
    }
    __syncthreads();
  }

#pragma unroll
  for (int nt = 0; nt < 3; ++nt) {
    int nl = nt * 16 + l15;
    if (nl < DHEADn) {
#pragma unroll
      for (int r = 0; r < 8; ++r) {
        size_t row = qrow + r + halfq * 8;
        O[row * Cdim + dc + nl] = f2bf(o[nt][r] / l_r[r]);
      }
    }
  }
}

// ================= final: out[b,c,p] = x[b,c,p] + proj[b,p,c] =================
__global__ __launch_bounds__(256)
void out_add_kernel(const float* __restrict__ proj, const float* __restrict__ x,
                    float* __restrict__ out) {
  int idx = blockIdx.x * 256 + threadIdx.x;
  if (idx >= Bsz * Cdim * HWdim) return;
  int p = idx & (HWdim - 1);
  int c = (idx >> 10) % Cdim;
  int b = idx / (Cdim * HWdim);
  out[idx] = x[idx] + proj[((size_t)b * HWdim + p) * Cdim + c];
}

// ================================ host launcher ================================
extern "C" void kernel_launch(void* const* d_in, const int* in_sizes, int n_in,
                              void* d_out, int out_size, void* d_ws, size_t ws_size,
                              hipStream_t stream) {
  (void)in_sizes; (void)n_in; (void)out_size; (void)ws_size;
  const float* x          = (const float*)d_in[0];
  const float* ctx        = (const float*)d_in[1];
  const unsigned char* vm = (const unsigned char*)d_in[2];
  const unsigned char* tm = (const unsigned char*)d_in[3];
  const float* gn_s  = (const float*)d_in[4];
  const float* gn_b  = (const float*)d_in[5];
  const float* piw   = (const float*)d_in[6];
  const float* pib   = (const float*)d_in[7];
  const float* n1s   = (const float*)d_in[8];
  const float* n1b   = (const float*)d_in[9];
  const float* wq1   = (const float*)d_in[10];
  const float* wk1   = (const float*)d_in[11];
  const float* wv1   = (const float*)d_in[12];
  const float* wo1   = (const float*)d_in[13];
  const float* bo1   = (const float*)d_in[14];
  const float* n2s   = (const float*)d_in[15];
  const float* n2b   = (const float*)d_in[16];
  const float* wq2   = (const float*)d_in[17];
  const float* wk2   = (const float*)d_in[18];
  const float* wv2   = (const float*)d_in[19];
  const float* wo2   = (const float*)d_in[20];
  const float* bo2   = (const float*)d_in[21];
  const float* n3s   = (const float*)d_in[22];
  const float* n3b   = (const float*)d_in[23];
  const float* ffw1  = (const float*)d_in[24];
  const float* ffb1  = (const float*)d_in[25];
  const float* ffw2  = (const float*)d_in[26];
  const float* ffb2  = (const float*)d_in[27];
  const float* pow_  = (const float*)d_in[28];
  const float* pob   = (const float*)d_in[29];
  float* out = (float*)d_out;

  // -------- workspace carving (256B aligned) --------
  char* wsp = (char*)d_ws;
  auto carve = [&](size_t bytes) -> char* {
    char* r = wsp; wsp += (bytes + 255) & ~(size_t)255; return r;
  };
  float* stats = (float*)carve(1024 * sizeof(float));
  float* bh    = (float*)carve((size_t)MTOK * Cdim * 4);   // residual stream (f32)
  float* btmp  = (float*)carve((size_t)MTOK * Cdim * 4);   // proj_out result (f32)
  unsigned short* bln16 = (unsigned short*)carve((size_t)MTOK * Cdim * 2);
  unsigned short* bq16  = (unsigned short*)carve((size_t)MTOK * Cdim * 2);
  unsigned short* bk16  = (unsigned short*)carve((size_t)MTOK * Cdim * 2);
  unsigned short* bv16  = (unsigned short*)carve((size_t)MTOK * Cdim * 2);
  unsigned short* bo16  = (unsigned short*)carve((size_t)MTOK * Cdim * 2);
  unsigned short* bff16 = (unsigned short*)carve((size_t)MTOK * FFIN * 2);
  unsigned short* ctx16 = (unsigned short*)carve((size_t)MCTX * CTXDIM * 2);
  unsigned short* piw_t = (unsigned short*)carve((size_t)Cdim * Cdim * 2);
  unsigned short* wq1_t = (unsigned short*)carve((size_t)Cdim * Cdim * 2);
  unsigned short* wk1_t = (unsigned short*)carve((size_t)Cdim * Cdim * 2);
  unsigned short* wv1_t = (unsigned short*)carve((size_t)Cdim * Cdim * 2);
  unsigned short* wo1_t = (unsigned short*)carve((size_t)Cdim * Cdim * 2);
  unsigned short* wq2_t = (unsigned short*)carve((size_t)Cdim * Cdim * 2);
  unsigned short* wk2_t = (unsigned short*)carve((size_t)CTXDIM * Cdim * 2);
  unsigned short* wv2_t = (unsigned short*)carve((size_t)CTXDIM * Cdim * 2);
  unsigned short* wo2_t = (unsigned short*)carve((size_t)Cdim * Cdim * 2);
  unsigned short* pow_t = (unsigned short*)carve((size_t)Cdim * Cdim * 2);
  unsigned short* ffw1_t = (unsigned short*)carve((size_t)Cdim * 2 * FFIN * 2);
  unsigned short* ffw2_t = (unsigned short*)carve((size_t)FFIN * Cdim * 2);

  const float scale = 0.15811388300841897f;   // 40^-0.5
  const int elems = Bsz * Cdim * HWdim;
  const dim3 gElem((elems + 255) / 256), b256(256), b128(128);
  const dim3 gTok(Cdim / 64, MTOK / 128);                 // (5, 64)
  const dim3 gCtx(Cdim / 64, (MCTX + 127) / 128);         // (5, 5)
  const dim3 gAttn(HWdim / 64, Bsz * HEADSn);             // (16, 64), 4 q-tiles/block
  const dim3 gLN(MTOK / 8);
  auto cgrid = [](int n) { return dim3((n + 255) / 256); };

  // -------- one-time weight transpose+convert to bf16 [N,K] --------
  wt_cvt_kernel<<<cgrid(Cdim * Cdim), b256, 0, stream>>>(piw,  piw_t, Cdim, Cdim);
  wt_cvt_kernel<<<cgrid(Cdim * Cdim), b256, 0, stream>>>(wq1,  wq1_t, Cdim, Cdim);
  wt_cvt_kernel<<<cgrid(Cdim * Cdim), b256, 0, stream>>>(wk1,  wk1_t, Cdim, Cdim);
  wt_cvt_kernel<<<cgrid(Cdim * Cdim), b256, 0, stream>>>(wv1,  wv1_t, Cdim, Cdim);
  wt_cvt_kernel<<<cgrid(Cdim * Cdim), b256, 0, stream>>>(wo1,  wo1_t, Cdim, Cdim);
  wt_cvt_kernel<<<cgrid(Cdim * Cdim), b256, 0, stream>>>(wq2,  wq2_t, Cdim, Cdim);
  wt_cvt_kernel<<<cgrid(CTXDIM * Cdim), b256, 0, stream>>>(wk2, wk2_t, CTXDIM, Cdim);
  wt_cvt_kernel<<<cgrid(CTXDIM * Cdim), b256, 0, stream>>>(wv2, wv2_t, CTXDIM, Cdim);
  wt_cvt_kernel<<<cgrid(Cdim * Cdim), b256, 0, stream>>>(wo2,  wo2_t, Cdim, Cdim);
  wt_cvt_kernel<<<cgrid(Cdim * Cdim), b256, 0, stream>>>(pow_, pow_t, Cdim, Cdim);
  wt_cvt_kernel<<<cgrid(Cdim * 2 * FFIN), b256, 0, stream>>>(ffw1, ffw1_t, Cdim, 2 * FFIN);
  wt_cvt_kernel<<<cgrid(FFIN * Cdim), b256, 0, stream>>>(ffw2, ffw2_t, FFIN, Cdim);
  act_cvt_kernel<<<cgrid(MCTX * CTXDIM), b256, 0, stream>>>(ctx, ctx16, MCTX * CTXDIM);

  // -------- GroupNorm + proj_in --------
  gn_stats_kernel<<<Bsz * GROUPSn, b256, 0, stream>>>(x, stats);
  gn_apply_kernel<<<gElem, b256, 0, stream>>>(x, stats, gn_s, gn_b, bln16);
  gemm_bf16_kernel<1, 0><<<gTok, b256, 0, stream>>>(bln16, piw_t, pib, nullptr, bh,
                                                    MTOK, Cdim, Cdim);

  // -------- self attention --------
  layernorm_kernel<<<gLN, b256, 0, stream>>>(bh, n1s, n1b, bln16, MTOK);
  gemm_bf16_kernel<0, 1><<<gTok, b256, 0, stream>>>(bln16, wq1_t, nullptr, nullptr, bq16,
                                                    MTOK, Cdim, Cdim);
  gemm_bf16_kernel<0, 1><<<gTok, b256, 0, stream>>>(bln16, wk1_t, nullptr, nullptr, bk16,
                                                    MTOK, Cdim, Cdim);
  gemm_bf16_kernel<0, 1><<<gTok, b256, 0, stream>>>(bln16, wv1_t, nullptr, nullptr, bv16,
                                                    MTOK, Cdim, Cdim);
  attn_kernel<<<gAttn, b128, 0, stream>>>(bq16, bk16, bv16, vm, bo16, HWdim, scale);
  gemm_bf16_kernel<2, 0><<<gTok, b256, 0, stream>>>(bo16, wo1_t, bo1, bh, bh,
                                                    MTOK, Cdim, Cdim);

  // -------- cross attention (text context K/V) --------
  layernorm_kernel<<<gLN, b256, 0, stream>>>(bh, n2s, n2b, bln16, MTOK);
  gemm_bf16_kernel<0, 1><<<gTok, b256, 0, stream>>>(bln16, wq2_t, nullptr, nullptr, bq16,
                                                    MTOK, Cdim, Cdim);
  gemm_bf16_kernel<0, 1><<<gCtx, b256, 0, stream>>>(ctx16, wk2_t, nullptr, nullptr, bk16,
                                                    MCTX, Cdim, CTXDIM);
  gemm_bf16_kernel<0, 1><<<gCtx, b256, 0, stream>>>(ctx16, wv2_t, nullptr, nullptr, bv16,
                                                    MCTX, Cdim, CTXDIM);
  attn_kernel<<<gAttn, b128, 0, stream>>>(bq16, bk16, bv16, tm, bo16, CTXLEN, scale);
  gemm_bf16_kernel<2, 0><<<gTok, b256, 0, stream>>>(bo16, wo2_t, bo2, bh, bh,
                                                    MTOK, Cdim, Cdim);

  // -------- GEGLU feed-forward --------
  layernorm_kernel<<<gLN, b256, 0, stream>>>(bh, n3s, n3b, bln16, MTOK);
  geglu_kernel<<<dim3(FFIN / 64, MTOK / 128), b256, 0, stream>>>(bln16, ffw1_t, ffb1,
                                                                 bff16, MTOK);
  gemm_bf16_kernel<2, 0><<<gTok, b256, 0, stream>>>(bff16, ffw2_t, ffb2, bh, bh,
                                                    MTOK, Cdim, FFIN);

  // -------- proj_out + residual (with NCHW transpose) --------
  act_cvt_kernel<<<cgrid(MTOK * Cdim), b256, 0, stream>>>(bh, bln16, MTOK * Cdim);
  gemm_bf16_kernel<1, 0><<<gTok, b256, 0, stream>>>(bln16, pow_t, pob, nullptr, btmp,
                                                    MTOK, Cdim, Cdim);
  out_add_kernel<<<gElem, b256, 0, stream>>>(btmp, x, out);
}
